// MultiHeadLatentAttention_24343874633994
// MI455X (gfx1250) — matrix operand order repaired and checked by reading, not compiled, not run on hardware
//
#include <hip/hip_runtime.h>

// ---------------- problem constants ----------------
#define BB    2
#define SS    2048
#define DIM   2048
#define HH_   16
#define QLR   1536
#define KVLR  512
#define NOPE  128
#define ROPE  64
#define VH    128
#define QKH   192          // NOPE + ROPE
#define MTOK  (BB*SS)      // 4096 tokens

typedef __attribute__((ext_vector_type(16))) __bf16 v16bf;
typedef __attribute__((ext_vector_type(8)))  float  v8f;

union BF16Frag { v16bf v; unsigned int u[8]; unsigned short h[16]; };
union F32Acc   { v8f   v; float f[8]; };

__device__ __forceinline__ unsigned short f2bf(float f) {
    unsigned int u = __float_as_uint(f);
    unsigned int r = u + 0x7fffu + ((u >> 16) & 1u);
    return (unsigned short)(r >> 16);
}

// load 16 bf16 elements (two 16B runs) into a fragment
__device__ __forceinline__ void load_16bf(BF16Frag& fr,
                                          const unsigned short* p0,
                                          const unsigned short* p1) {
    uint4 a = *(const uint4*)p0;
    uint4 b = *(const uint4*)p1;
    fr.u[0] = a.x; fr.u[1] = a.y; fr.u[2] = a.z; fr.u[3] = a.w;
    fr.u[4] = b.x; fr.u[5] = b.y; fr.u[6] = b.z; fr.u[7] = b.w;
}

__device__ __forceinline__ v8f wmma_bf16(const BF16Frag& a, const BF16Frag& b, v8f c) {
    return __builtin_amdgcn_wmma_f32_16x16x32_bf16(
        false, a.v, false, b.v, (short)0, c, false, false);
}

// async global -> LDS copy of 16 bytes (ASYNCcnt-tracked). Low 32 bits of a
// generic shared pointer are the LDS byte offset (shared aperture is in the
// high 32 bits per the CDNA5 aperture check).
__device__ __forceinline__ void async_copy_b128(const unsigned short* gsrc,
                                                unsigned short* lds_dst) {
    unsigned int loff = (unsigned int)(unsigned long long)lds_dst;
    asm volatile("global_load_async_to_lds_b128 %0, %1, off"
                 :: "v"(loff), "v"(gsrc) : "memory");
}

__device__ __forceinline__ float redmax16(float v) {
    v = fmaxf(v, __shfl_xor(v, 1, 32));
    v = fmaxf(v, __shfl_xor(v, 2, 32));
    v = fmaxf(v, __shfl_xor(v, 4, 32));
    v = fmaxf(v, __shfl_xor(v, 8, 32));
    return v;
}
__device__ __forceinline__ float redsum16(float v) {
    v += __shfl_xor(v, 1, 32);
    v += __shfl_xor(v, 2, 32);
    v += __shfl_xor(v, 4, 32);
    v += __shfl_xor(v, 8, 32);
    return v;
}

// ---------------- elementwise kernels ----------------
__global__ void k_cvt_bf16(const float* __restrict__ in, unsigned short* __restrict__ out, int n) {
    int i = blockIdx.x * 256 + threadIdx.x;
    if (i < n) out[i] = f2bf(in[i]);
}

// in [K,N] fp32  ->  out [N,K] bf16 (transposed for B-fragment contiguity)
__global__ void k_cvt_t(const float* __restrict__ in, unsigned short* __restrict__ out, int K, int N) {
    int i = blockIdx.x * 256 + threadIdx.x;
    if (i >= K * N) return;
    int k = i / N, n = i - k * N;
    out[(size_t)n * K + k] = f2bf(in[i]);
}

// RMSNorm over first C cols of [rows, stride] fp32 -> [rows, C] bf16
__global__ void k_rmsnorm(const float* __restrict__ in, int stride, int C,
                          const float* __restrict__ w, unsigned short* __restrict__ out) {
    __shared__ float red[256];
    int row = blockIdx.x;
    const float* p = in + (size_t)row * stride;
    float s = 0.f;
    for (int c = threadIdx.x; c < C; c += 256) { float v = p[c]; s += v * v; }
    red[threadIdx.x] = s;
    __syncthreads();
    for (int off = 128; off; off >>= 1) {
        if ((int)threadIdx.x < off) red[threadIdx.x] += red[threadIdx.x + off];
        __syncthreads();
    }
    float inv = rsqrtf(red[0] / (float)C + 1e-6f);
    for (int c = threadIdx.x; c < C; c += 256)
        out[(size_t)row * C + c] = f2bf(p[c] * inv * w[c]);
}

// q [4096,3072] fp32 -> qf [tok][h][192] bf16, RoPE on last 64 of each head
__global__ void k_rope_q(const float* __restrict__ q, unsigned short* __restrict__ qf) {
    long long i = (long long)blockIdx.x * 256 + threadIdx.x;
    if (i >= (long long)MTOK * HH_ * QKH) return;
    int d  = (int)(i % QKH);
    int th = (int)((i / QKH) % HH_);
    int tok = (int)(i / ((long long)QKH * HH_));
    int pos = tok & (SS - 1);
    const float* qrow = q + (size_t)tok * (HH_ * QKH) + th * QKH;
    float val;
    if (d < NOPE) {
        val = qrow[d];
    } else {
        int j = d - NOPE;
        int hi = (j >= 32) ? 1 : 0;
        int fi = hi ? j - 32 : j;
        float freq = __powf(10000.0f, -(float)fi / 32.0f);
        float ang = (float)pos * freq;
        float c = __cosf(ang), sn = __sinf(ang);
        float t1 = qrow[NOPE + fi], t2 = qrow[NOPE + 32 + fi];
        val = hi ? (t2 * c + t1 * sn) : (t1 * c - t2 * sn);
    }
    qf[i] = f2bf(val);
}

// kf [tok][h][192]: 0..127 from kvu k_nope, 128..191 RoPE'd k_rope (broadcast over h)
__global__ void k_pack_k(const float* __restrict__ kvu, const float* __restrict__ kvd,
                         unsigned short* __restrict__ kf) {
    long long i = (long long)blockIdx.x * 256 + threadIdx.x;
    if (i >= (long long)MTOK * HH_ * QKH) return;
    int d  = (int)(i % QKH);
    int th = (int)((i / QKH) % HH_);
    int tok = (int)(i / ((long long)QKH * HH_));
    int pos = tok & (SS - 1);
    float val;
    if (d < NOPE) {
        val = kvu[(size_t)tok * (HH_ * (NOPE + VH)) + th * (NOPE + VH) + d];
    } else {
        int j = d - NOPE;
        int hi = (j >= 32) ? 1 : 0;
        int fi = hi ? j - 32 : j;
        float freq = __powf(10000.0f, -(float)fi / 32.0f);
        float ang = (float)pos * freq;
        float c = __cosf(ang), sn = __sinf(ang);
        const float* kr = kvd + (size_t)tok * (KVLR + ROPE) + KVLR;
        float t1 = kr[fi], t2 = kr[32 + fi];
        val = hi ? (t2 * c + t1 * sn) : (t1 * c - t2 * sn);
    }
    kf[i] = f2bf(val);
}

// vt [b][h][vd][s] bf16  (transposed so PV B-fragments are contiguous along keys)
__global__ void k_pack_v(const float* __restrict__ kvu, unsigned short* __restrict__ vt) {
    long long i = (long long)blockIdx.x * 256 + threadIdx.x;
    if (i >= (long long)MTOK * HH_ * VH) return;
    int vd = (int)(i % VH);
    int th = (int)((i / VH) % HH_);
    int tok = (int)(i / ((long long)VH * HH_));
    int b = tok >> 11, s = tok & (SS - 1);
    vt[((((size_t)b * HH_ + th) * VH) + vd) * SS + s] =
        f2bf(kvu[(size_t)tok * (HH_ * (NOPE + VH)) + th * (NOPE + VH) + NOPE + vd]);
}

// ---------------- bf16 WMMA GEMM, 64x64 block tile ----------------
// C[M,N] fp32 = A[M,K]bf16 @ Bt[N,K]bf16^T + bias[N]
// 4 waves/block; each wave owns a 16x64 strip (1 A-frag reused by 4 WMMAs).
// B panel (64 cols x 32 k) is async-staged into double-buffered LDS.
// K-loop unrolled by 2 with ping-pong buffers/registers (K % 64 == 0 for all
// call sites), removing fragment copies and per-iteration wait branches.
__global__ void __launch_bounds__(128) k_gemm(const unsigned short* __restrict__ A,
                                              const unsigned short* __restrict__ Bt,
                                              const float* __restrict__ bias,
                                              float* __restrict__ C,
                                              int M, int N, int K) {
    __shared__ unsigned short bs[2][64 * 32];   // 2 x 4KB B panels

    const int t    = threadIdx.x;
    const int lane = t & 31;
    const int wave = t >> 5;
    const int row0 = blockIdx.y * 64 + wave * 16;
    const int col0 = blockIdx.x * 64;
    const int m16 = lane & 15;
    const int hh  = lane >> 4;

    // B staging mapping: 128 threads cover 64 rows x (2 halves of 32 k)
    const int nl   = t >> 1;        // 0..63 : local column (row of Bt)
    const int part = t & 1;         // 0..1  : which 16-element half of k-step
    const unsigned short* bsrc = Bt + (size_t)(col0 + nl) * K + part * 16;
    unsigned short* l0 = &bs[0][nl * 32 + part * 16];
    unsigned short* l1 = &bs[1][nl * 32 + part * 16];

    F32Acc acc[4];
#pragma unroll
    for (int nt = 0; nt < 4; ++nt)
#pragma unroll
        for (int r = 0; r < 8; ++r) acc[nt].f[r] = 0.f;

    const unsigned short* pa0 = A + (size_t)(row0 + m16) * K + hh * 16;

    // prologue: stage k-step 0 into buffer 0, preload A frag 0
    async_copy_b128(bsrc, l0);
    async_copy_b128(bsrc + 8, l0 + 8);
    BF16Frag af0, af1;
    load_16bf(af0, pa0, pa0 + 8);

    for (int k0 = 0; k0 < K; k0 += 64) {
        // ---- half 1: compute with buf0/af0, stage buf1 (k0+32), preload af1 ----
        async_copy_b128(bsrc + k0 + 32, l1);
        async_copy_b128(bsrc + k0 + 40, l1 + 8);
        load_16bf(af1, pa0 + k0 + 32, pa0 + k0 + 40);
        __builtin_prefetch(pa0 + k0 + 256, 0, 1);   // global_prefetch_b8 (speculative)
        asm volatile("s_wait_asynccnt 0x2" ::: "memory");   // buf0 complete
        __syncthreads();
#pragma unroll
        for (int nt = 0; nt < 4; ++nt) {
            BF16Frag bf;
            const unsigned short* p = &bs[0][(nt * 16 + m16) * 32 + hh * 16];
            load_16bf(bf, p, p + 8);
            acc[nt].v = wmma_bf16(af0, bf, acc[nt].v);
        }
        __syncthreads();   // buf0 consumed by all waves

        // ---- half 2: compute with buf1/af1, stage buf0 (k0+64) if more ----
        const bool more = (k0 + 64) < K;
        if (more) {
            async_copy_b128(bsrc + k0 + 64, l0);
            async_copy_b128(bsrc + k0 + 72, l0 + 8);
            load_16bf(af0, pa0 + k0 + 64, pa0 + k0 + 72);
            asm volatile("s_wait_asynccnt 0x2" ::: "memory");   // buf1 complete
        } else {
            asm volatile("s_wait_asynccnt 0x0" ::: "memory");
        }
        __syncthreads();
#pragma unroll
        for (int nt = 0; nt < 4; ++nt) {
            BF16Frag bf;
            const unsigned short* p = &bs[1][(nt * 16 + m16) * 32 + hh * 16];
            load_16bf(bf, p, p + 8);
            acc[nt].v = wmma_bf16(af1, bf, acc[nt].v);
        }
        __syncthreads();   // buf1 consumed by all waves
    }

#pragma unroll
    for (int nt = 0; nt < 4; ++nt) {
        float b = bias[col0 + nt * 16 + m16];
#pragma unroll
        for (int r = 0; r < 8; ++r) {
            int row = row0 + r + 8 * hh;
            C[(size_t)row * N + col0 + nt * 16 + m16] = acc[nt].f[r] + b;
        }
    }
}

// ---------------- flash attention (causal, online softmax) ----------------
__global__ void __launch_bounds__(128) k_attn(const unsigned short* __restrict__ qf,
                                              const unsigned short* __restrict__ kf,
                                              const unsigned short* __restrict__ vt,
                                              unsigned short* __restrict__ obuf) {
    __shared__ unsigned short pbuf[4][16 * 32];
    const int lane = threadIdx.x & 31;
    const int wave = threadIdx.x >> 5;
    const int q0 = (blockIdx.x * 4 + wave) * 16;
    const int h  = blockIdx.y;
    const int b  = blockIdx.z;
    const int m16 = lane & 15;
    const int hh  = lane >> 4;

    // Q tile: 6 A-fragments of 16x32 bf16 covering d = 0..191
    BF16Frag qa[6];
    {
        const unsigned short* qr = qf + (((size_t)(b * SS + q0 + m16)) * HH_ + h) * QKH;
#pragma unroll
        for (int kb = 0; kb < 6; ++kb) {
            const unsigned short* p = qr + kb * 32 + hh * 16;
            load_16bf(qa[kb], p, p + 8);
        }
    }

    float rowM[8], rowL[8];
    F32Acc Oc[8];
#pragma unroll
    for (int r = 0; r < 8; ++r) { rowM[r] = -3.0e38f; rowL[r] = 0.f; }
#pragma unroll
    for (int n0 = 0; n0 < 8; ++n0)
#pragma unroll
        for (int r = 0; r < 8; ++r) Oc[n0].f[r] = 0.f;

    const float scale = 0.072168783648703329f; // 1/sqrt(192)
    unsigned short* lds = &pbuf[wave][0];

    for (int j0 = 0; j0 <= q0 + 15; j0 += 32) {
        F32Acc sc[2];
#pragma unroll
        for (int tt = 0; tt < 2; ++tt) {
            const int jb = j0 + tt * 16;
#pragma unroll
            for (int r = 0; r < 8; ++r) sc[tt].f[r] = 0.f;
            const unsigned short* kr = kf + (((size_t)(b * SS + jb + m16)) * HH_ + h) * QKH;
#pragma unroll
            for (int kb = 0; kb < 6; ++kb) {
                BF16Frag bfr;
                const unsigned short* p = kr + kb * 32 + hh * 16;
                load_16bf(bfr, p, p + 8);
                sc[tt].v = wmma_bf16(qa[kb], bfr, sc[tt].v);
            }
#pragma unroll
            for (int r = 0; r < 8; ++r) {
                int qrow = q0 + r + 8 * hh;
                int key  = jb + m16;
                float v = sc[tt].f[r] * scale;
                sc[tt].f[r] = (key <= qrow) ? v : -3.0e38f;
            }
        }
        // online softmax state update
        float corr[8], mnew[8];
#pragma unroll
        for (int r = 0; r < 8; ++r) {
            float rm = redmax16(fmaxf(sc[0].f[r], sc[1].f[r]));
            float mn = fmaxf(rowM[r], rm);
            mnew[r] = mn;
            corr[r] = __expf(rowM[r] - mn);
        }
#pragma unroll
        for (int tt = 0; tt < 2; ++tt)
#pragma unroll
            for (int r = 0; r < 8; ++r)
                sc[tt].f[r] = __expf(sc[tt].f[r] - mnew[r]);
#pragma unroll
        for (int r = 0; r < 8; ++r) {
            float ls = redsum16(sc[0].f[r] + sc[1].f[r]);
            rowL[r] = rowL[r] * corr[r] + ls;
            rowM[r] = mnew[r];
        }
#pragma unroll
        for (int n0 = 0; n0 < 8; ++n0)
#pragma unroll
            for (int r = 0; r < 8; ++r) Oc[n0].f[r] *= corr[r];

        // stage P (16x32 bf16, row-major) in per-wave LDS to re-layout as A-fragment
#pragma unroll
        for (int tt = 0; tt < 2; ++tt)
#pragma unroll
            for (int r = 0; r < 8; ++r) {
                int row = r + 8 * hh;
                lds[row * 32 + tt * 16 + m16] = f2bf(sc[tt].f[r]);
            }
        asm volatile("s_wait_dscnt 0" ::: "memory");   // LDS stores visible wave-wide
        BF16Frag pa;
        {
            const unsigned short* p = lds + m16 * 32 + hh * 16;
            load_16bf(pa, p, p + 8);
        }
        asm volatile("s_wait_dscnt 0" ::: "memory");

        // PV: 8 fragments cover v-dims 0..127
#pragma unroll
        for (int n0 = 0; n0 < 8; ++n0) {
            BF16Frag vb;
            const unsigned short* p =
                vt + ((((size_t)b * HH_ + h) * VH) + n0 * 16 + m16) * SS + j0 + hh * 16;
            load_16bf(vb, p, p + 8);
            Oc[n0].v = wmma_bf16(pa, vb, Oc[n0].v);
        }
    }

    float invL[8];
#pragma unroll
    for (int r = 0; r < 8; ++r) invL[r] = 1.0f / rowL[r];
#pragma unroll
    for (int n0 = 0; n0 < 8; ++n0)
#pragma unroll
        for (int r = 0; r < 8; ++r) {
            size_t row = (size_t)(b * SS + q0 + r + 8 * hh);
            obuf[row * (HH_ * VH) + h * VH + n0 * 16 + m16] = f2bf(Oc[n0].f[r] * invL[r]);
        }
}

// ---------------- host launch ----------------
extern "C" void kernel_launch(void* const* d_in, const int* in_sizes, int n_in,
                              void* d_out, int out_size, void* d_ws, size_t ws_size,
                              hipStream_t stream) {
    const float* x        = (const float*)d_in[0];
    const float* wqd      = (const float*)d_in[1];
    const float* bqd      = (const float*)d_in[2];
    const float* qnorm_w  = (const float*)d_in[3];
    const float* wqu      = (const float*)d_in[4];
    const float* bqu      = (const float*)d_in[5];
    const float* wkvd     = (const float*)d_in[6];
    const float* bkvd     = (const float*)d_in[7];
    const float* kvnorm_w = (const float*)d_in[8];
    const float* wkvu     = (const float*)d_in[9];
    const float* bkvu     = (const float*)d_in[10];
    const float* wout     = (const float*)d_in[11];
    const float* bout     = (const float*)d_in[12];

    size_t off = 0;
    auto carve = [&](size_t bytes) -> void* {
        void* p = (char*)d_ws + off;
        off += (bytes + 255) & ~(size_t)255;
        return p;
    };
    unsigned short* xb    = (unsigned short*)carve((size_t)MTOK * DIM * 2);
    unsigned short* wqdT  = (unsigned short*)carve((size_t)QLR * DIM * 2);
    unsigned short* wquT  = (unsigned short*)carve((size_t)HH_ * QKH * QLR * 2);
    unsigned short* wkvdT = (unsigned short*)carve((size_t)(KVLR + ROPE) * DIM * 2);
    unsigned short* wkvuT = (unsigned short*)carve((size_t)HH_ * (NOPE + VH) * KVLR * 2);
    unsigned short* woutT = (unsigned short*)carve((size_t)DIM * (HH_ * VH) * 2);
    float*          qlat  = (float*)carve((size_t)MTOK * QLR * 4);
    unsigned short* qlatn = (unsigned short*)carve((size_t)MTOK * QLR * 2);
    float*          kvdb  = (float*)carve((size_t)MTOK * (KVLR + ROPE) * 4);
    unsigned short* kvn   = (unsigned short*)carve((size_t)MTOK * KVLR * 2);
    float*          qfull = (float*)carve((size_t)MTOK * HH_ * QKH * 4);
    float*          kvu   = (float*)carve((size_t)MTOK * HH_ * (NOPE + VH) * 4);
    unsigned short* qfb   = (unsigned short*)carve((size_t)MTOK * HH_ * QKH * 2);
    unsigned short* kfb   = (unsigned short*)carve((size_t)MTOK * HH_ * QKH * 2);
    unsigned short* vtb   = (unsigned short*)carve((size_t)MTOK * HH_ * VH * 2);
    unsigned short* obuf  = (unsigned short*)carve((size_t)MTOK * HH_ * VH * 2);

    auto cdiv = [](long long a, long long b) { return (int)((a + b - 1) / b); };

    // convert x + transpose-convert all weights to bf16
    k_cvt_bf16<<<cdiv((long long)MTOK * DIM, 256), 256, 0, stream>>>(x, xb, MTOK * DIM);
    k_cvt_t<<<cdiv((long long)DIM * QLR, 256), 256, 0, stream>>>(wqd, wqdT, DIM, QLR);
    k_cvt_t<<<cdiv((long long)QLR * HH_ * QKH, 256), 256, 0, stream>>>(wqu, wquT, QLR, HH_ * QKH);
    k_cvt_t<<<cdiv((long long)DIM * (KVLR + ROPE), 256), 256, 0, stream>>>(wkvd, wkvdT, DIM, KVLR + ROPE);
    k_cvt_t<<<cdiv((long long)KVLR * HH_ * (NOPE + VH), 256), 256, 0, stream>>>(wkvu, wkvuT, KVLR, HH_ * (NOPE + VH));
    k_cvt_t<<<cdiv((long long)(HH_ * VH) * DIM, 256), 256, 0, stream>>>(wout, woutT, HH_ * VH, DIM);

    // down projections (N: 1536, 576 -> both divisible by 64)
    k_gemm<<<dim3(QLR / 64, MTOK / 64), 128, 0, stream>>>(xb, wqdT, bqd, qlat, MTOK, QLR, DIM);
    k_gemm<<<dim3((KVLR + ROPE) / 64, MTOK / 64), 128, 0, stream>>>(xb, wkvdT, bkvd, kvdb, MTOK, KVLR + ROPE, DIM);

    // rmsnorms -> bf16
    k_rmsnorm<<<MTOK, 256, 0, stream>>>(qlat, QLR, QLR, qnorm_w, qlatn);
    k_rmsnorm<<<MTOK, 256, 0, stream>>>(kvdb, KVLR + ROPE, KVLR, kvnorm_w, kvn);

    // up projections
    k_gemm<<<dim3(HH_ * QKH / 64, MTOK / 64), 128, 0, stream>>>(qlatn, wquT, bqu, qfull, MTOK, HH_ * QKH, QLR);
    k_gemm<<<dim3(HH_ * (NOPE + VH) / 64, MTOK / 64), 128, 0, stream>>>(kvn, wkvuT, bkvu, kvu, MTOK, HH_ * (NOPE + VH), KVLR);

    // rope + pack q/k/v in bf16 (v transposed)
    k_rope_q<<<cdiv((long long)MTOK * HH_ * QKH, 256), 256, 0, stream>>>(qfull, qfb);
    k_pack_k<<<cdiv((long long)MTOK * HH_ * QKH, 256), 256, 0, stream>>>(kvu, kvdb, kfb);
    k_pack_v<<<cdiv((long long)MTOK * HH_ * VH, 256), 256, 0, stream>>>(kvu, vtb);

    // causal flash attention
    k_attn<<<dim3(SS / 64, HH_, BB), 128, 0, stream>>>(qfb, kfb, vtb, obuf);

    // output projection -> d_out (fp32)
    k_gemm<<<dim3(DIM / 64, MTOK / 64), 128, 0, stream>>>(obuf, woutT, bout, (float*)d_out, MTOK, DIM, HH_ * VH);
}
